// ClassAtt_74491912782221
// MI455X (gfx1250) — matrix operand order repaired and checked by reading, not compiled, not run on hardware
//
#include <hip/hip_runtime.h>
#include <hip/hip_bf16.h>

// MoE-ish classifier pipeline on MI455X (gfx1250, wave32):
// all GEMMs via v_wmma_f32_16x16x32_bf16. 128x128x64 block tiles, 8 waves,
// each wave computes a 32x64 macro-tile (2x4 WMMA tiles, 16 WMMA per LDS tile).
// fp32 operands are converted to bf16 with packed cvt during global->LDS staging;
// register prefetch + global_prefetch_b8 overlap VMEM with WMMA issue.

typedef __bf16 bf16_t;
typedef __attribute__((ext_vector_type(16))) __bf16 v16bf;
typedef __attribute__((ext_vector_type(8)))  __bf16 v8bf;
typedef __attribute__((ext_vector_type(2)))  __bf16 v2bf;
typedef __attribute__((ext_vector_type(2)))  float  v2f;
typedef __attribute__((ext_vector_type(8)))  float  v8f;

#define BM 128
#define BN 128
#define BK 64
#define SA 72   // LDS row stride (bf16 elements) for A tile (64 + 8 pad, 16B aligned)
#define SB 72   // LDS row stride for B tile

__device__ __forceinline__ bf16_t f2bf(float f) { return (bf16_t)f; }

// Packed fp32 pair -> bf16 pair (one v_cvt_pk_bf16_f32, no half-reg fixups)
__device__ __forceinline__ unsigned int pk2(float a, float b) {
    v2f f = {a, b};
    v2bf h = __builtin_convertvector(f, v2bf);
    unsigned int r;
    __builtin_memcpy(&r, &h, 4);
    return r;
}
__device__ __forceinline__ uint4 cvt8(const float* __restrict__ s) {
    return make_uint4(pk2(s[0], s[1]), pk2(s[2], s[3]), pk2(s[4], s[5]), pk2(s[6], s[7]));
}

// A-fragment (16x32 bf16, MxK): lane holds row M=lane%16;
// half-wave g=lane/16: elems 0..7 = K[8g..8g+7], elems 8..15 = K[16+8g..16+8g+7]
__device__ __forceinline__ v16bf frag_a(const bf16_t* tile, int lane) {
    const int row = lane & 15, g = lane >> 4;
    v8bf lo = *(const v8bf*)(tile + row * SA + g * 8);
    v8bf hi = *(const v8bf*)(tile + row * SA + 16 + g * 8);
    return __builtin_shufflevector(lo, hi, 0,1,2,3,4,5,6,7,8,9,10,11,12,13,14,15);
}

// B-fragment (32x16 bf16, KxN) built from row-major W rows (W[n][k]):
// lane holds col N=lane%16; half-wave g: elems 0..7 = K[8g..8g+7], elems 8..15 = K[16+8g..16+8g+7]
__device__ __forceinline__ v16bf frag_b(const bf16_t* tile, int lane) {
    const int col = lane & 15, g = lane >> 4;
    v8bf lo = *(const v8bf*)(tile + col * SB + g * 8);
    v8bf hi = *(const v8bf*)(tile + col * SB + 16 + g * 8);
    return __builtin_shufflevector(lo, hi, 0,1,2,3,4,5,6,7,8,9,10,11,12,13,14,15);
}

// C = act(A(MxK) @ W(NxK)^T + bias). A fp32 or bf16 (AF32); W/bias fp32.
// Output bf16 (intermediate) or fp32 (final); ReLU optional.
// M multiple of 128, K multiple of 64; N ragged (guarded) for the 1000-col layer.
template <bool AF32, bool RELU, bool OUTF32>
__global__ void __launch_bounds__(256)
gemm_wmma_bf16(const void* __restrict__ Aptr, long lda,
               const float* __restrict__ W, long ldw,
               const float* __restrict__ bias,
               void* __restrict__ Cptr, long ldc,
               int N, int K) {
    __shared__ __align__(16) bf16_t lA[BM * SA];
    __shared__ __align__(16) bf16_t lB[BN * SB];

    const int t    = threadIdx.x;
    const int lane = t & 31;
    const int wave = t >> 5;
    const int wm   = wave & 3;   // 4 waves along M (32 rows each)
    const int wn   = wave >> 2;  // 2 waves along N (64 cols each)
    const long m0  = (long)blockIdx.y * BM;
    const long n0  = (long)blockIdx.x * BN;

    // Global->LDS staging: each thread owns 32 consecutive elems of one row
    const int arow = t >> 1, ahalf = (t & 1) * 32;  // A: 128 x 64
    const int brow = t >> 1, bhalf = (t & 1) * 32;  // B: 128 x 64

    v8f acc[2][4] = {};
    uint4 ra[4], rb[4];

    auto loadA = [&](int k0) {
        if (AF32) {
            const float* src = (const float*)Aptr + (m0 + arow) * lda + k0 + ahalf;
#pragma unroll
            for (int i = 0; i < 4; ++i) ra[i] = cvt8(src + 8 * i);
        } else {
            const uint4* src = (const uint4*)((const bf16_t*)Aptr + (m0 + arow) * lda + k0 + ahalf);
#pragma unroll
            for (int i = 0; i < 4; ++i) ra[i] = src[i];
        }
    };
    auto loadB = [&](int k0) {
        const long r = n0 + brow;
        if (r < (long)N) {
            const float* src = W + r * ldw + k0 + bhalf;
#pragma unroll
            for (int i = 0; i < 4; ++i) rb[i] = cvt8(src + 8 * i);
        } else {
#pragma unroll
            for (int i = 0; i < 4; ++i) rb[i] = make_uint4(0u, 0u, 0u, 0u);
        }
    };

    loadA(0);
    loadB(0);

    const int KT = K / BK;
    for (int kt = 0; kt < KT; ++kt) {
        __syncthreads();  // previous iteration done reading LDS
#pragma unroll
        for (int i = 0; i < 4; ++i) {
            *(uint4*)(lA + arow * SA + ahalf + 8 * i) = ra[i];
            *(uint4*)(lB + brow * SB + bhalf + 8 * i) = rb[i];
        }
        __syncthreads();  // tile visible

        if (kt + 1 < KT) {            // overlap next tile's VMEM with WMMA issue
            loadA((kt + 1) * BK);
            loadB((kt + 1) * BK);
        }
        if (kt + 2 < KT) {            // deep prefetch into GL2 (global_prefetch_b8)
            __builtin_prefetch((const char*)W + ((n0 + brow) * ldw + (long)(kt + 2) * BK + bhalf) * sizeof(float), 0, 1);
            if (AF32)
                __builtin_prefetch((const float*)Aptr + (m0 + arow) * lda + (kt + 2) * BK + ahalf, 0, 1);
            else
                __builtin_prefetch((const bf16_t*)Aptr + (m0 + arow) * lda + (kt + 2) * BK + ahalf, 0, 1);
        }

#pragma unroll
        for (int kk = 0; kk < 2; ++kk) {  // two WMMA K-steps per LDS tile
            const int ko = kk * 32;
            v16bf a0 = frag_a(lA + (wm * 32)      * SA + ko, lane);
            v16bf a1 = frag_a(lA + (wm * 32 + 16) * SA + ko, lane);
            v16bf b0 = frag_b(lB + (wn * 64)      * SB + ko, lane);
            v16bf b1 = frag_b(lB + (wn * 64 + 16) * SB + ko, lane);
            v16bf b2 = frag_b(lB + (wn * 64 + 32) * SB + ko, lane);
            v16bf b3 = frag_b(lB + (wn * 64 + 48) * SB + ko, lane);

            acc[0][0] = __builtin_amdgcn_wmma_f32_16x16x32_bf16(false, a0, false, b0, (short)0, acc[0][0], false, false);
            acc[0][1] = __builtin_amdgcn_wmma_f32_16x16x32_bf16(false, a0, false, b1, (short)0, acc[0][1], false, false);
            acc[0][2] = __builtin_amdgcn_wmma_f32_16x16x32_bf16(false, a0, false, b2, (short)0, acc[0][2], false, false);
            acc[0][3] = __builtin_amdgcn_wmma_f32_16x16x32_bf16(false, a0, false, b3, (short)0, acc[0][3], false, false);
            acc[1][0] = __builtin_amdgcn_wmma_f32_16x16x32_bf16(false, a1, false, b0, (short)0, acc[1][0], false, false);
            acc[1][1] = __builtin_amdgcn_wmma_f32_16x16x32_bf16(false, a1, false, b1, (short)0, acc[1][1], false, false);
            acc[1][2] = __builtin_amdgcn_wmma_f32_16x16x32_bf16(false, a1, false, b2, (short)0, acc[1][2], false, false);
            acc[1][3] = __builtin_amdgcn_wmma_f32_16x16x32_bf16(false, a1, false, b3, (short)0, acc[1][3], false, false);
        }
    }

    // C/D layout: lane holds col N=lane%16; vgpr i holds row i + 8*(lane/16)
    const int cn = lane & 15, g = lane >> 4;
#pragma unroll
    for (int tm = 0; tm < 2; ++tm)
#pragma unroll
        for (int tn = 0; tn < 4; ++tn)
#pragma unroll
            for (int i = 0; i < 8; ++i) {
                const long gr = m0 + wm * 32 + tm * 16 + g * 8 + i;
                const long gc = n0 + wn * 64 + tn * 16 + cn;
                if (gc < (long)N) {
                    float v = acc[tm][tn][i] + bias[gc];
                    if (RELU) v = fmaxf(v, 0.0f);
                    if (OUTF32) ((float*)Cptr)[gr * ldc + gc] = v;
                    else        ((bf16_t*)Cptr)[gr * ldc + gc] = f2bf(v);
                }
            }
}

// Per-row attention: alpha_e = <last_hid, P_e>, softmax over 3, context into D[:,0:1024].
// P: (B x 3072) bf16; D: (B x 2048) bf16 with cols 1024..2047 = last_hid (already written).
__global__ void __launch_bounds__(256)
attn_context_kernel(const bf16_t* __restrict__ P, bf16_t* __restrict__ D) {
    const long row = blockIdx.x;
    const int  t    = threadIdx.x;
    const int  lane = t & 31;
    const int  wave = t >> 5;
    const bf16_t* p = P + row * 3072;
    bf16_t*       d = D + row * 2048;

    float s1 = 0.f, s2 = 0.f, s3 = 0.f;
    for (int c = t; c < 1024; c += 256) {
        const float h = (float)d[1024 + c];
        s1 += h * (float)p[c];
        s2 += h * (float)p[1024 + c];
        s3 += h * (float)p[2048 + c];
    }
    // wave32 butterfly reduction
#pragma unroll
    for (int off = 16; off > 0; off >>= 1) {
        s1 += __shfl_xor(s1, off, 32);
        s2 += __shfl_xor(s2, off, 32);
        s3 += __shfl_xor(s3, off, 32);
    }
    __shared__ float r1[8], r2[8], r3[8];
    if (lane == 0) { r1[wave] = s1; r2[wave] = s2; r3[wave] = s3; }
    __syncthreads();
    __shared__ float wsh[3];
    if (t == 0) {
        float a1 = 0.f, a2 = 0.f, a3 = 0.f;
#pragma unroll
        for (int i = 0; i < 8; ++i) { a1 += r1[i]; a2 += r2[i]; a3 += r3[i]; }
        const float m  = fmaxf(a1, fmaxf(a2, a3));
        const float e1 = __expf(a1 - m), e2 = __expf(a2 - m), e3 = __expf(a3 - m);
        const float inv = 1.0f / (e1 + e2 + e3);
        wsh[0] = e1 * inv; wsh[1] = e2 * inv; wsh[2] = e3 * inv;
    }
    __syncthreads();
    const float w1 = wsh[0], w2 = wsh[1], w3 = wsh[2];
    for (int c = t; c < 1024; c += 256) {
        const float ctx = w1 * (float)p[c] + w2 * (float)p[1024 + c] + w3 * (float)p[2048 + c];
        d[c] = f2bf(ctx);
    }
}

extern "C" void kernel_launch(void* const* d_in, const int* in_sizes, int n_in,
                              void* d_out, int out_size, void* d_ws, size_t ws_size,
                              hipStream_t stream) {
    (void)in_sizes; (void)n_in; (void)out_size; (void)ws_size;
    const int B = 16384;

    const float* tube  = (const float*)d_in[0];
    const float* wW[3] = {(const float*)d_in[1], (const float*)d_in[3], (const float*)d_in[5]};
    const float* wb[3] = {(const float*)d_in[2], (const float*)d_in[4], (const float*)d_in[6]};
    const float* whW   = (const float*)d_in[7];
    const float* whb   = (const float*)d_in[8];
    const float* wd1W  = (const float*)d_in[9];
    const float* wd1b  = (const float*)d_in[10];
    const float* wd2W  = (const float*)d_in[11];
    const float* wd2b  = (const float*)d_in[12];

    // Workspace: P (B x 3072 bf16), D=decode_start (B x 2048 bf16), H (B x 2048 bf16)
    bf16_t* P = (bf16_t*)d_ws;
    bf16_t* D = P + (size_t)B * 3072;
    bf16_t* H = D + (size_t)B * 2048;

    const dim3 blk(256);
    const dim3 gridM(1024 / BN, B / BM);                 // N=1024
    const dim3 gridD(2048 / BN, B / BM);                 // N=2048
    const dim3 gridO((1000 + BN - 1) / BN, B / BM);      // N=1000 (guarded)

    // Stage 1: three expert GEMMs, fp32 A (tube slices), ReLU, bf16 out into P columns
    for (int e = 0; e < 3; ++e) {
        gemm_wmma_bf16<true, true, false><<<gridM, blk, 0, stream>>>(
            tube + e * 512, 1536L, wW[e], 512L, wb[e], P + e * 1024, 3072L, 1024, 512);
    }
    // Stage 2: last_hid = relu(P @ wh^T + b) into D[:,1024:2048]
    gemm_wmma_bf16<false, true, false><<<gridM, blk, 0, stream>>>(
        P, 3072L, whW, 3072L, whb, D + 1024, 2048L, 1024, 3072);
    // Stage 3: attention weights + context into D[:,0:1024]
    attn_context_kernel<<<dim3(B), blk, 0, stream>>>(P, D);
    // Stage 4: H = relu(D @ wd1^T + b)
    gemm_wmma_bf16<false, true, false><<<gridD, blk, 0, stream>>>(
        D, 2048L, wd1W, 2048L, wd1b, H, 2048L, 2048, 2048);
    // Stage 5: out = H @ wd2^T + b (fp32 output, N=1000 guarded)
    gemm_wmma_bf16<false, false, true><<<gridO, blk, 0, stream>>>(
        H, 2048L, wd2W, 2048L, wd2b, d_out, 1000L, 1000, 2048);
}
